// LSTM_model_84756884619955
// MI455X (gfx1250) — matrix operand order repaired and checked by reading, not compiled, not run on hardware
//
#include <hip/hip_runtime.h>
#include <hip/hip_bf16.h>

typedef __attribute__((ext_vector_type(16))) __bf16 v16bf;
typedef __attribute__((ext_vector_type(8)))  __bf16 v8bf;
typedef __attribute__((ext_vector_type(8)))  float  v8f;

#define B_   32
#define S_   512
#define F_   512
#define H_   1024
#define O_   512
#define G4H  4096     // 4*H
#define KTOT 1536     // H + F (fused [h|x] @ [U;W])
#define NWG  64

// ---------------- WMMA fragment loaders (CDNA5 wave32 layouts) ----------------
// A (16x32 bf16, row-major source): lane%16 = row; lanes 0-15 hold K {0..7,16..23},
// lanes 16-31 hold K {8..15,24..31}.
static __device__ __forceinline__ v16bf frag_A(const __bf16* rowptr, int k, int half) {
  v8bf lo = *(const v8bf*)(rowptr + k + half * 8);
  v8bf hi = *(const v8bf*)(rowptr + k + 16 + half * 8);
  v16bf r;
#pragma unroll
  for (int i = 0; i < 8; ++i) { r[i] = lo[i]; r[i + 8] = hi[i]; }
  return r;
}
// B (32x16 bf16) from an N-major (transposed) weight matrix: lane%16 = column N,
// lanes 0-15 hold K {0..7,16..23}, lanes 16-31 hold K {8..15,24..31}.
static __device__ __forceinline__ v16bf frag_B(const __bf16* rowptr, int k, int half) {
  v8bf lo = *(const v8bf*)(rowptr + k + half * 8);
  v8bf hi = *(const v8bf*)(rowptr + k + 16 + half * 8);
  v16bf r;
#pragma unroll
  for (int i = 0; i < 8; ++i) { r[i] = lo[i]; r[i + 8] = hi[i]; }
  return r;
}

// Fast transcendental path: v_exp_f32 + v_rcp_f32, no IEEE division sequence.
static __device__ __forceinline__ float sigm(float x) {
  return __builtin_amdgcn_rcpf(1.f + __expf(-x));
}
static __device__ __forceinline__ float tanh_fast(float x) {
  // tanh(x) = 1 - 2/(e^{2x}+1)
  float r = __builtin_amdgcn_rcpf(__expf(2.f * x) + 1.f);
  return __builtin_fmaf(-2.f, r, 1.f);
}

// ---------------- pre-pass: pack/transpose to bf16 ----------------
__global__ void pack_x(const float* __restrict__ x, __bf16* __restrict__ xbf) {
  int idx = blockIdx.x * 256 + threadIdx.x;          // < S*B*F = 8388608
  int f = idx & (F_ - 1);
  int r = idx >> 9;                                   // r = s*32 + b
  int b = r & 31;
  int s = r >> 5;
  xbf[idx] = (__bf16)x[((size_t)b * S_ + s) * F_ + f];
}

__global__ void pack_uw(const float* __restrict__ U, const float* __restrict__ W,
                        __bf16* __restrict__ uwt) {
  int idx = blockIdx.x * 256 + threadIdx.x;          // < 4096*1536
  int k = idx % KTOT;
  int n = idx / KTOT;
  float v = (k < H_) ? U[(size_t)k * G4H + n] : W[(size_t)(k - H_) * G4H + n];
  uwt[idx] = (__bf16)v;
}

__global__ void pack_v(const float* __restrict__ V, __bf16* __restrict__ vt) {
  int idx = blockIdx.x * 256 + threadIdx.x;          // < 512*1024
  int k = idx & (H_ - 1);
  int o = idx >> 10;
  vt[idx] = (__bf16)V[(size_t)k * O_ + o];
}

__global__ void init_state(__bf16* __restrict__ hbuf, unsigned* __restrict__ cnt) {
  int idx = blockIdx.x * 256 + threadIdx.x;
  if (idx < 2 * B_ * H_) hbuf[idx] = (__bf16)0.f;
  if (idx == 0) *cnt = 0u;
}

// ---------------- persistent recurrent kernel ----------------
// 64 WGs x 256 threads. WG wg owns H-columns [wg*16, wg*16+16).
// Wave w: gate g = w&3, row-tile mt = w>>2. Gates GEMM: [h|x](32x1536) @ UWt tile.
__global__ __launch_bounds__(256, 1)
void lstm_recurrent(const __bf16* __restrict__ xbf,
                    const __bf16* __restrict__ uwt,
                    const float*  __restrict__ hid_bias,
                    __bf16* __restrict__ hbuf,   // 2 * [32][1024] double buffer
                    __bf16* __restrict__ hsbf,   // [B*S][H] bf16 copy for out-proj
                    float*  __restrict__ out,    // full d_out
                    unsigned* __restrict__ cnt) {
  __shared__ float gates[4][B_][16];

  const int tid  = threadIdx.x;
  const int wg   = blockIdx.x;         // 0..63
  const int wv   = tid >> 5;
  const int lane = tid & 31;
  const int half = lane >> 4;
  const int ln   = lane & 15;
  const int g    = wv & 3;
  const int mt   = wv >> 2;

  const int n0 = g * H_ + wg * 16;     // column block in 4H space
  const __bf16* btrow = uwt + (size_t)(n0 + ln) * KTOT;   // K-major row for column n0+ln

  float creg[2] = {0.f, 0.f};          // persistent cell state (2 elems/thread)

  const size_t OUT_OFF = (size_t)B_ * S_ * H_;
  const size_t CS_OFF  = OUT_OFF + (size_t)B_ * S_ * O_;
  const size_t FS_OFF  = CS_OFF + (size_t)S_ * B_ * H_;
  const size_t OS_OFF  = FS_OFF + (size_t)S_ * B_ * H_;
  const size_t IS_OFF  = OS_OFF + (size_t)S_ * B_ * H_;

  for (int s = 0; s < S_; ++s) {
    const __bf16* hprev = hbuf + (size_t)(s & 1) * (B_ * H_);
    __bf16*       hnext = hbuf + (size_t)((s + 1) & 1) * (B_ * H_);

    const __bf16* arow_h = hprev + (size_t)(mt * 16 + ln) * H_;
    const __bf16* arow_x = xbf + (size_t)(s * B_ + mt * 16 + ln) * F_;

    // Two accumulators -> two independent WMMA dependency chains.
    v8f acc0 = {}, acc1 = {};

    // h-part: K = 0..1023 (affine addresses, no selects)
#pragma unroll 4
    for (int kt = 0; kt < H_ / 32; kt += 2) {
      v16bf a0 = frag_A(arow_h, kt * 32, half);
      v16bf b0 = frag_B(btrow, kt * 32, half);
      acc0 = __builtin_amdgcn_wmma_f32_16x16x32_bf16(false, a0, false, b0,
                                                     (short)0, acc0, false, false);
      v16bf a1 = frag_A(arow_h, kt * 32 + 32, half);
      v16bf b1 = frag_B(btrow, kt * 32 + 32, half);
      acc1 = __builtin_amdgcn_wmma_f32_16x16x32_bf16(false, a1, false, b1,
                                                     (short)0, acc1, false, false);
    }
    // x-part: K = 1024..1535
    const __bf16* btrow_x = btrow + H_;
#pragma unroll 4
    for (int kt = 0; kt < F_ / 32; kt += 2) {
      v16bf a0 = frag_A(arow_x, kt * 32, half);
      v16bf b0 = frag_B(btrow_x, kt * 32, half);
      acc0 = __builtin_amdgcn_wmma_f32_16x16x32_bf16(false, a0, false, b0,
                                                     (short)0, acc0, false, false);
      v16bf a1 = frag_A(arow_x, kt * 32 + 32, half);
      v16bf b1 = frag_B(btrow_x, kt * 32 + 32, half);
      acc1 = __builtin_amdgcn_wmma_f32_16x16x32_bf16(false, a1, false, b1,
                                                     (short)0, acc1, false, false);
    }

    // Merge chains + write gate tile. D layout: elem v -> (row = v + 8*half, col = ln)
#pragma unroll
    for (int v = 0; v < 8; ++v)
      gates[g][mt * 16 + v + 8 * half][ln] = acc0[v] + acc1[v];
    __syncthreads();

    // Elementwise cell update: 512 elems, 2 per thread.
#pragma unroll
    for (int e = 0; e < 2; ++e) {
      int idx = tid * 2 + e;
      int row = idx >> 4;              // batch index
      int col = idx & 15;
      int colg = wg * 16 + col;

      float gi = gates[0][row][col] + hid_bias[colg];
      float gf = gates[1][row][col] + hid_bias[H_ + colg];
      float gg = gates[2][row][col] + hid_bias[2 * H_ + colg];
      float go = gates[3][row][col] + hid_bias[3 * H_ + colg];

      float iv = sigm(gi);
      float fv = sigm(gf);
      float gv = tanh_fast(gg);
      float ov = sigm(go);
      float cv = fv * creg[e] + iv * gv;
      creg[e] = cv;
      float hv = ov * tanh_fast(cv);

      hnext[(size_t)row * H_ + colg] = (__bf16)hv;
      hsbf[((size_t)row * S_ + s) * H_ + colg] = (__bf16)hv;
      out[(size_t)row * (S_ * H_) + (size_t)s * H_ + colg] = hv;  // hidden_seq [B,S,H]

      size_t gidx = (size_t)s * (B_ * H_) + (size_t)row * H_ + colg;  // [S,B,H]
      out[CS_OFF + gidx] = cv;
      out[FS_OFF + gidx] = fv;
      out[OS_OFF + gidx] = ov;
      out[IS_OFF + gidx] = iv;
    }

    // Grid-wide barrier before next step consumes hnext.
    __syncthreads();
    if (tid == 0) {
      __hip_atomic_fetch_add(cnt, 1u, __ATOMIC_ACQ_REL, __HIP_MEMORY_SCOPE_AGENT);
      const unsigned target = (unsigned)(NWG * (s + 1));
      while (__hip_atomic_load(cnt, __ATOMIC_ACQUIRE, __HIP_MEMORY_SCOPE_AGENT) < target) {
        __builtin_amdgcn_s_sleep(1);
      }
    }
    __syncthreads();
  }
}

// ---------------- output projection: hidden[B*S,H] @ V[H,O] + bias ----------------
// 1x4 register tiling: each wave computes one 16-row tile x four 16-col tiles,
// reusing every A fragment across 4 WMMAs.
__global__ __launch_bounds__(256)
void out_proj(const __bf16* __restrict__ hsbf, const __bf16* __restrict__ vt,
              const float* __restrict__ out_bias, float* __restrict__ out) {
  const int tid  = threadIdx.x;
  const int wv   = tid >> 5;
  const int lane = tid & 31;
  const int half = lane >> 4;
  const int ln   = lane & 15;

  int wgl   = blockIdx.x * 8 + wv;     // 0..8191 wave-tiles
  int ng    = wgl & 7;                 // N-group: 8 groups of 4 tiles (O/16 = 32)
  int tileM = wgl >> 3;                // 0..1023

  const __bf16* arow = hsbf + (size_t)(tileM * 16 + ln) * H_;
  const __bf16* brow[4];
#pragma unroll
  for (int j = 0; j < 4; ++j)
    brow[j] = vt + (size_t)((ng * 4 + j) * 16 + ln) * H_;

  v8f acc[4] = {};
#pragma unroll 2
  for (int kt = 0; kt < H_ / 32; ++kt) {
    v16bf a = frag_A(arow, kt * 32, half);
#pragma unroll
    for (int j = 0; j < 4; ++j) {
      v16bf b = frag_B(brow[j], kt * 32, half);
      acc[j] = __builtin_amdgcn_wmma_f32_16x16x32_bf16(false, a, false, b,
                                                       (short)0, acc[j], false, false);
    }
  }

  float* obase = out + (size_t)B_ * S_ * H_;   // OUT_OFF
#pragma unroll
  for (int j = 0; j < 4; ++j) {
    int n = (ng * 4 + j) * 16 + ln;
    float ob = out_bias[n];
#pragma unroll
    for (int v = 0; v < 8; ++v) {
      int row = tileM * 16 + v + 8 * half;
      obase[(size_t)row * O_ + n] = acc[j][v] + ob;
    }
  }
}

// ---------------- launch ----------------
extern "C" void kernel_launch(void* const* d_in, const int* in_sizes, int n_in,
                              void* d_out, int out_size, void* d_ws, size_t ws_size,
                              hipStream_t stream) {
  const float* x        = (const float*)d_in[0];
  const float* W        = (const float*)d_in[1];
  const float* U        = (const float*)d_in[2];
  const float* hid_bias = (const float*)d_in[3];
  const float* V        = (const float*)d_in[4];
  const float* out_bias = (const float*)d_in[5];
  float* out = (float*)d_out;

  // Workspace layout (bytes): [cnt 256][hbuf 128K][xbf 16M][uwt 12M][vt 1M][hsbf 32M]
  char* base = (char*)d_ws;
  unsigned* cnt = (unsigned*)base;
  __bf16* hbuf = (__bf16*)(base + 256);
  __bf16* xbf  = (__bf16*)(base + 256 + 2 * B_ * H_ * sizeof(__bf16));
  __bf16* uwt  = xbf + (size_t)S_ * B_ * F_;        // 8388608 elems
  __bf16* vt   = uwt + (size_t)G4H * KTOT;          // 6291456 elems
  __bf16* hsbf = vt + (size_t)O_ * H_;              // 524288 elems

  init_state<<<256, 256, 0, stream>>>(hbuf, cnt);
  pack_x<<<(S_ * B_ * F_) / 256, 256, 0, stream>>>(x, xbf);
  pack_uw<<<(G4H * KTOT) / 256, 256, 0, stream>>>(U, W, uwt);
  pack_v<<<(O_ * H_) / 256, 256, 0, stream>>>(V, vt);

  lstm_recurrent<<<NWG, 256, 0, stream>>>(xbf, uwt, hid_bias, hbuf, hsbf, out, cnt);

  // waves = 1024 M-tiles * 8 N-groups = 8192 ; 8 waves per block -> 1024 blocks
  out_proj<<<1024, 256, 0, stream>>>(hsbf, vt, out_bias, out);
}
